// SingleHeadCausalAttention_11982958756139
// MI455X (gfx1250) — compile-verified, matrix operand order checked by reading, not run on hardware
//
#include <hip/hip_runtime.h>

// ---------------------------------------------------------------------------
// Types for CDNA5 WMMA (wave32): v_wmma_f32_16x16x32_bf16
// ---------------------------------------------------------------------------
typedef __bf16 bf16_t;
typedef __attribute__((ext_vector_type(16))) __bf16 v16bf;
typedef __attribute__((ext_vector_type(8)))  __bf16 v8bf;   // 16-byte chunk
typedef __attribute__((ext_vector_type(8)))  float  v8f;

constexpr int BATCH = 4, SEQ = 4096, DMODEL = 1024, DKDIM = 128, FFNDIM = 2048;
constexpr float LN_EPS = 1e-5f;

// A/B fragment (16-bit, 16x32): lane l holds row/col (l&15); lanes<16 carry
// K = {0..7, 16..23}, lanes>=16 carry K = {8..15, 24..31}. Caller passes
// base = &row[kb] with kb = (lane<16)?0:8; we read base[0..7] and base[16..23].
__device__ __forceinline__ v16bf load_frag(const bf16_t* base) {
  union { v16bf v; v8bf h[2]; } u;
  u.h[0] = *(const v8bf*)(base);
  u.h[1] = *(const v8bf*)(base + 16);
  return u.v;
}

__device__ __forceinline__ v8f wmma_bf16(v16bf a, v16bf b, v8f c) {
  // (neg_a, A, neg_b, B, c_mod, C, reuse_a, reuse_b)
  return __builtin_amdgcn_wmma_f32_16x16x32_bf16(false, a, false, b, (short)0, c,
                                                 false, false);
}

// ---------------------------------------------------------------------------
// CDNA5 async global -> LDS copy (ASYNCcnt-tracked). Each active lane copies
// 16 bytes. LDS destination offset = low 32 bits of a generic pointer to
// __shared__ memory. Inline asm: portable across ROCm/amdgpu toolchains.
// ---------------------------------------------------------------------------
__device__ __forceinline__ void async_copy_b128(void* lds_dst, const void* gsrc) {
  unsigned int lds_off = (unsigned int)(unsigned long long)lds_dst;
  asm volatile("global_load_async_to_lds_b128 %0, %1, off"
               :
               : "v"(lds_off), "v"(gsrc)
               : "memory");
}
__device__ __forceinline__ void wait_async0() {
  asm volatile("s_wait_asynccnt 0x0" ::: "memory");
}

// ---------------------------------------------------------------------------
// fp32 -> bf16 elementwise (weights)
// ---------------------------------------------------------------------------
__global__ void cvt_f32_bf16_kernel(const float* __restrict__ in,
                                    bf16_t* __restrict__ out, int n) {
  int i = blockIdx.x * blockDim.x + threadIdx.x;
  if (i < n) out[i] = (bf16_t)in[i];
}

// ---------------------------------------------------------------------------
// Row LayerNorm: fp32 in -> bf16 out. One block per row, 256 threads, D=1024.
// ---------------------------------------------------------------------------
__global__ __launch_bounds__(256) void layernorm_bf16_kernel(
    const float* __restrict__ x, const float* __restrict__ g,
    const float* __restrict__ b, bf16_t* __restrict__ out) {
  __shared__ float red[256];
  const size_t row = blockIdx.x;
  const float* xr = x + row * DMODEL;
  const int t = threadIdx.x;
  float v[4]; float s = 0.f;
  for (int i = 0; i < 4; ++i) { v[i] = xr[t + i * 256]; s += v[i]; }
  red[t] = s; __syncthreads();
  for (int o = 128; o > 0; o >>= 1) { if (t < o) red[t] += red[t + o]; __syncthreads(); }
  const float mu = red[0] * (1.f / DMODEL);
  __syncthreads();
  float vs = 0.f;
  for (int i = 0; i < 4; ++i) { float d = v[i] - mu; vs += d * d; }
  red[t] = vs; __syncthreads();
  for (int o = 128; o > 0; o >>= 1) { if (t < o) red[t] += red[t + o]; __syncthreads(); }
  const float rstd = rsqrtf(red[0] * (1.f / DMODEL) + LN_EPS);
  for (int i = 0; i < 4; ++i) {
    int c = t + i * 256;
    out[row * DMODEL + c] = (bf16_t)((v[i] - mu) * rstd * g[c] + b[c]);
  }
}

// ---------------------------------------------------------------------------
// bf16 transpose: [B,T,D] -> [B,D,T] (so ctx = P*h has row-major B fragments)
// ---------------------------------------------------------------------------
__global__ void transpose_bf16_kernel(const bf16_t* __restrict__ in,
                                      bf16_t* __restrict__ out) {
  __shared__ bf16_t tile[32][33];
  const int b = blockIdx.z, t0 = blockIdx.x * 32, d0 = blockIdx.y * 32;
  for (int j = threadIdx.y; j < 32; j += 8)
    tile[j][threadIdx.x] = in[((size_t)b * SEQ + t0 + j) * DMODEL + d0 + threadIdx.x];
  __syncthreads();
  for (int j = threadIdx.y; j < 32; j += 8)
    out[((size_t)b * DMODEL + d0 + j) * SEQ + t0 + threadIdx.x] = tile[threadIdx.x][j];
}

// ---------------------------------------------------------------------------
// Generic bf16 WMMA GEMM: out = [relu]( A[M,K] * W[N,K]^T + bias + resid )
// Block tile 64x128 (8 waves, each 32x32 via 2x2 WMMA 16x16x32), K step 32.
// Tiles move via GLOBAL_LOAD_ASYNC_TO_LDS_B128 with double-buffered LDS:
// stage(k+1) is in flight (ASYNCcnt) while WMMAs consume tile k.
// ---------------------------------------------------------------------------
__global__ __launch_bounds__(256) void gemm_bf16_kernel(
    const bf16_t* __restrict__ A, const bf16_t* __restrict__ W,
    const float* __restrict__ bias, const float* __restrict__ resid,
    float* __restrict__ outF, bf16_t* __restrict__ outB,
    int M, int N, int K, int relu) {
  constexpr int BM = 64, BN = 128, BK = 32;
  __shared__ __align__(16) bf16_t As[2][BM * BK];
  __shared__ __align__(16) bf16_t Bs[2][BN * BK];
  const int tid = threadIdx.x, wave = tid >> 5, lane = tid & 31;
  const int wm = wave & 1, wn = wave >> 1;                  // 2 x 4 wave grid
  const int m0 = blockIdx.y * BM, n0 = blockIdx.x * BN;
  const int nl = lane & 15;
  const int kb = (lane < 16) ? 0 : 8;
  const int rbase = (lane < 16) ? 0 : 8;
  const int lr = tid >> 2, lc = (tid & 3) * 8;              // 16B chunk per lane
  v8f acc[2][2] = {};

  auto stage = [&](int buf, int kk) {
    async_copy_b128(&As[buf][lr * BK + lc],        &A[(size_t)(m0 + lr) * K + kk + lc]);
    async_copy_b128(&Bs[buf][lr * BK + lc],        &W[(size_t)(n0 + lr) * K + kk + lc]);
    async_copy_b128(&Bs[buf][(lr + 64) * BK + lc], &W[(size_t)(n0 + lr + 64) * K + kk + lc]);
  };

  stage(0, 0);
  wait_async0();
  __syncthreads();

  int buf = 0;
  for (int kk = 0; kk < K; kk += BK) {
    if (kk + BK < K) stage(buf ^ 1, kk + BK);   // async copy of next tile
    v16bf af[2], bfr[2];
    for (int am = 0; am < 2; ++am)
      af[am] = load_frag(&As[buf][(wm * 32 + am * 16 + nl) * BK + kb]);
    for (int bn = 0; bn < 2; ++bn)
      bfr[bn] = load_frag(&Bs[buf][(wn * 32 + bn * 16 + nl) * BK + kb]);
    for (int am = 0; am < 2; ++am)
      for (int bn = 0; bn < 2; ++bn)
        acc[am][bn] = wmma_bf16(af[am], bfr[bn], acc[am][bn]);
    wait_async0();        // my async stores to LDS landed
    __syncthreads();      // everyone's landed; everyone done reading buf
    buf ^= 1;
  }

  for (int am = 0; am < 2; ++am)
    for (int bn = 0; bn < 2; ++bn)
      for (int r = 0; r < 8; ++r) {
        int gm = m0 + wm * 32 + am * 16 + rbase + r;
        int gn = n0 + wn * 32 + bn * 16 + nl;
        float v = acc[am][bn][r];
        if (bias)  v += bias[gn];
        if (resid) v += resid[(size_t)gm * N + gn];
        if (relu)  v = fmaxf(v, 0.f);
        size_t o = (size_t)gm * N + gn;
        if (outF) outF[o] = v;
        if (outB) outB[o] = (bf16_t)v;
      }
}

// ---------------------------------------------------------------------------
// Flash attention with causal mask + ALiBi (slope 1), attends over h (bf16).
// One block = 16 query rows x full D. 8 waves; wave0 computes S=q*k^T tiles
// (16x32) and publishes P + rescale factors via LDS; every wave accumulates a
// 16x128 slice of ctx = P * h^T. Online softmax, nothing T x T hits HBM.
// ---------------------------------------------------------------------------
__global__ __launch_bounds__(256) void attn_kernel(
    const bf16_t* __restrict__ q, const bf16_t* __restrict__ kmat,
    const bf16_t* __restrict__ hT, bf16_t* __restrict__ ctx) {
  __shared__ __align__(16) bf16_t Pld[16 * 32];
  __shared__ float alphaLds[16];
  __shared__ float linvLds[16];
  const int b = blockIdx.y;
  const int t0 = blockIdx.x * 16;
  const int tid = threadIdx.x, wave = tid >> 5, lane = tid & 31;
  const int nl = lane & 15;
  const int kb = (lane < 16) ? 0 : 8;
  const int rbase = (lane < 16) ? 0 : 8;
  const float scale = 0.08838834764831845f;  // 1/sqrt(128)

  v8f acc[8] = {};                 // rows t0..t0+15, cols wave*128 + nt*16 + nl
  v16bf qf[4];
  float mi[8], li[8];
  if (wave == 0) {
    const bf16_t* qrow = q + ((size_t)b * SEQ + t0 + nl) * DKDIM + kb;
    for (int kk = 0; kk < 4; ++kk) qf[kk] = load_frag(qrow + kk * 32);
    for (int r = 0; r < 8; ++r) { mi[r] = -1e30f; li[r] = 0.f; }
  }

  const int nIter = (t0 + 16 + 31) / 32;
  for (int it = 0; it < nIter; ++it) {
    const int s0 = it * 32;
    if (wave == 0) {
      v8f S[2] = {};
      for (int ns = 0; ns < 2; ++ns) {
        const bf16_t* krow = kmat + ((size_t)b * SEQ + s0 + ns * 16 + nl) * DKDIM + kb;
        for (int kk = 0; kk < 4; ++kk)
          S[ns] = wmma_bf16(qf[kk], load_frag(krow + kk * 32), S[ns]);
      }
      for (int r = 0; r < 8; ++r) {
        const int trow = t0 + rbase + r;
        const int c0 = s0 + nl, c1 = s0 + 16 + nl;
        float v0 = (c0 <= trow) ? S[0][r] * scale - (float)(trow - c0) : -1e30f;
        float v1 = (c1 <= trow) ? S[1][r] * scale - (float)(trow - c1) : -1e30f;
        float rmax = fmaxf(v0, v1);
        for (int o = 1; o < 16; o <<= 1) rmax = fmaxf(rmax, __shfl_xor(rmax, o, 32));
        const float nm = fmaxf(mi[r], rmax);
        const float p0 = __expf(v0 - nm), p1 = __expf(v1 - nm);
        float rs = p0 + p1;
        for (int o = 1; o < 16; o <<= 1) rs += __shfl_xor(rs, o, 32);
        const float al = __expf(mi[r] - nm);
        li[r] = li[r] * al + rs;
        mi[r] = nm;
        Pld[(rbase + r) * 32 + nl]      = (bf16_t)p0;
        Pld[(rbase + r) * 32 + 16 + nl] = (bf16_t)p1;
        if (nl == 0) alphaLds[rbase + r] = al;
      }
    }
    __syncthreads();  // P + alpha visible to all waves
    float al[8];
    for (int r = 0; r < 8; ++r) al[r] = alphaLds[rbase + r];
    const v16bf pf = load_frag(&Pld[nl * 32 + kb]);   // A fragment of P (16x32)
    for (int nt = 0; nt < 8; ++nt) {
      for (int r = 0; r < 8; ++r) acc[nt][r] *= al[r];
      const int c = wave * 128 + nt * 16 + nl;
      const bf16_t* hrow = hT + ((size_t)b * DMODEL + c) * SEQ + s0 + kb;
      acc[nt] = wmma_bf16(pf, load_frag(hrow), acc[nt]);
    }
    __syncthreads();  // before wave0 overwrites Pld next iteration
  }

  if (wave == 0 && nl == 0)
    for (int r = 0; r < 8; ++r) linvLds[rbase + r] = 1.f / li[r];
  __syncthreads();
  float linv[8];
  for (int r = 0; r < 8; ++r) linv[r] = linvLds[rbase + r];
  for (int nt = 0; nt < 8; ++nt) {
    const int c = wave * 128 + nt * 16 + nl;
    for (int r = 0; r < 8; ++r)
      ctx[((size_t)b * SEQ + t0 + rbase + r) * DMODEL + c] = (bf16_t)(acc[nt][r] * linv[r]);
  }
}

// ---------------------------------------------------------------------------
// Host-side orchestration
// ---------------------------------------------------------------------------
extern "C" void kernel_launch(void* const* d_in, const int* in_sizes, int n_in,
                              void* d_out, int out_size, void* d_ws, size_t ws_size,
                              hipStream_t stream) {
  const float* x    = (const float*)d_in[0];
  const float* ln1g = (const float*)d_in[1];
  const float* ln1b = (const float*)d_in[2];
  const float* Wq   = (const float*)d_in[3];
  const float* Wk   = (const float*)d_in[4];
  const float* Wv   = (const float*)d_in[5];
  const float* ln2g = (const float*)d_in[6];
  const float* ln2b = (const float*)d_in[7];
  const float* W1   = (const float*)d_in[8];
  const float* b1   = (const float*)d_in[9];
  const float* W2   = (const float*)d_in[10];
  const float* b2   = (const float*)d_in[11];

  const size_t MT = (size_t)BATCH * SEQ;  // 16384 rows
  char* ws = (char*)d_ws;
  size_t off = 0;
  auto take = [&](size_t bytes) -> char* {
    char* p = ws + off;
    off += (bytes + 255) & ~(size_t)255;
    return p;
  };
  bf16_t* hb   = (bf16_t*)take(MT * DMODEL * 2);   // h (LN1 out), reused as h2
  bf16_t* hT   = (bf16_t*)take(MT * DMODEL * 2);   // h transposed [B,D,T]
  bf16_t* qb   = (bf16_t*)take(MT * DKDIM * 2);
  bf16_t* kbuf = (bf16_t*)take(MT * DKDIM * 2);
  bf16_t* ctxb = (bf16_t*)take(MT * DMODEL * 2);
  float*  x1   = (float*)take(MT * DMODEL * 4);    // x + ctx*Wv^T
  bf16_t* ffb  = (bf16_t*)take(MT * FFNDIM * 2);
  bf16_t* Wqb  = (bf16_t*)take((size_t)DKDIM * DMODEL * 2);
  bf16_t* Wkb  = (bf16_t*)take((size_t)DKDIM * DMODEL * 2);
  bf16_t* Wvb  = (bf16_t*)take((size_t)DMODEL * DMODEL * 2);
  bf16_t* W1b  = (bf16_t*)take((size_t)FFNDIM * DMODEL * 2);
  bf16_t* W2b  = (bf16_t*)take((size_t)DMODEL * FFNDIM * 2);
  (void)in_sizes; (void)n_in; (void)out_size; (void)ws_size;

  auto cvt = [&](const float* src, bf16_t* dst, int n) {
    cvt_f32_bf16_kernel<<<(n + 255) / 256, 256, 0, stream>>>(src, dst, n);
  };
  cvt(Wq, Wqb, DKDIM * DMODEL);
  cvt(Wk, Wkb, DKDIM * DMODEL);
  cvt(Wv, Wvb, DMODEL * DMODEL);
  cvt(W1, W1b, FFNDIM * DMODEL);
  cvt(W2, W2b, DMODEL * FFNDIM);

  // h = LN1(x) in bf16; hT for the P*h matmul
  layernorm_bf16_kernel<<<(int)MT, 256, 0, stream>>>(x, ln1g, ln1b, hb);
  {
    dim3 g(SEQ / 32, DMODEL / 32, BATCH), blk(32, 8);
    transpose_bf16_kernel<<<g, blk, 0, stream>>>(hb, hT);
  }

  // q = h Wq^T, k = h Wk^T   (bf16 out)
  {
    dim3 g(DKDIM / 128, (int)(MT / 64));
    gemm_bf16_kernel<<<g, 256, 0, stream>>>(hb, Wqb, nullptr, nullptr, nullptr, qb,
                                            (int)MT, DKDIM, DMODEL, 0);
    gemm_bf16_kernel<<<g, 256, 0, stream>>>(hb, Wkb, nullptr, nullptr, nullptr, kbuf,
                                            (int)MT, DKDIM, DMODEL, 0);
  }

  // ctx = softmax(causal+alibi(q k^T / sqrt(dk))) * h
  {
    dim3 g(SEQ / 16, BATCH);
    attn_kernel<<<g, 256, 0, stream>>>(qb, kbuf, hT, ctxb);
  }

  // x1 = x + ctx Wv^T   (fp32)
  {
    dim3 g(DMODEL / 128, (int)(MT / 64));
    gemm_bf16_kernel<<<g, 256, 0, stream>>>(ctxb, Wvb, nullptr, x, x1, nullptr,
                                            (int)MT, DMODEL, DMODEL, 0);
  }

  // h2 = LN2(x1)  (reuse hb)
  layernorm_bf16_kernel<<<(int)MT, 256, 0, stream>>>(x1, ln2g, ln2b, hb);

  // ff = relu(h2 W1^T + b1) in bf16
  {
    dim3 g(FFNDIM / 128, (int)(MT / 64));
    gemm_bf16_kernel<<<g, 256, 0, stream>>>(hb, W1b, b1, nullptr, nullptr, ffb,
                                            (int)MT, FFNDIM, DMODEL, 1);
  }

  // out = x1 + ff W2^T + b2  (fp32, final output)
  {
    dim3 g(DMODEL / 128, (int)(MT / 64));
    gemm_bf16_kernel<<<g, 256, 0, stream>>>(ffb, W2b, b2, x1, (float*)d_out, nullptr,
                                            (int)MT, DMODEL, FFNDIM, 0);
  }
}
